// GNN_38087769981372
// MI455X (gfx1250) — matrix operand order repaired and checked by reading, not compiled, not run on hardware
//
#include <hip/hip_runtime.h>
#include <hip/hip_bf16.h>

// ---------------------------------------------------------------------------
// GNN forward for MI455X (gfx1250, wave32, WMMA, LDS-staged weights).
//   p1 = x + scatter_add(x[src] -> dst)            (fp32, atomics)
//   z1 = relu(p1 @ W1 + b1)                        (bf16 WMMA, fp32 acc)
//   p2 = z1 + scatter_add(z1[src] -> dst)
//   h  = p2 @ W2 + b2
//   pooled = segment_sum(h, idx, G=500)
//   zg = relu(pooled @ W3 + b3)
//   out = log_softmax(zg @ W4 + b4)
// GEMM: one wave owns a 16x128 output strip (32 v_wmma).  Weights staged in
// LDS once per workgroup (row stride padded to 136 bf16: 16B-aligned b128
// accesses, conflict-free phase-0 banks 4*m16).  A fragment loaded once per
// K-step and reused by 8 WMMAs; 8 B fragments prefetched per K-step.
// ---------------------------------------------------------------------------

typedef __attribute__((ext_vector_type(16))) __bf16 v16bf;
typedef __attribute__((ext_vector_type(8)))  __bf16 v8bf;
typedef __attribute__((ext_vector_type(8)))  float  v8f;

#define N_NODES  50000
#define N_EDGES  800000
#define DFEAT    128
#define N_GRAPHS 500
#define N_CLASS  16
#define LDS_STRIDE 136   // 128 + 8 bf16 pad: 272B rows -> 16B aligned, bank-safe

__device__ __forceinline__ __bf16 f2bf(float f) {
    unsigned u = __builtin_bit_cast(unsigned, f);
    u += 0x7FFFu + ((u >> 16) & 1u);              // round-to-nearest-even
    unsigned short h = (unsigned short)(u >> 16);
    return __builtin_bit_cast(__bf16, h);
}

// ---- transpose + fp32->bf16 convert a [128 x 128] weight: Wt[n][k] = W[k][n]
__global__ void gnn_prep_w(const float* __restrict__ W, __bf16* __restrict__ Wt) {
    int t = blockIdx.x * 256 + threadIdx.x;       // 16384 threads total
    int n = t >> 7, k = t & 127;
    Wt[n * DFEAT + k] = f2bf(W[k * DFEAT + n]);
}

// ---- fp32 -> bf16 activation convert, 8 elements per thread
__global__ void gnn_cvt_bf16(const float* __restrict__ src, __bf16* __restrict__ dst, int n8) {
    int i = blockIdx.x * blockDim.x + threadIdx.x;
    if (i >= n8) return;
    const float4* s = (const float4*)src + 2 * (size_t)i;
    float4 a = s[0], b = s[1];
    v8bf o;
    o[0] = f2bf(a.x); o[1] = f2bf(a.y); o[2] = f2bf(a.z); o[3] = f2bf(a.w);
    o[4] = f2bf(b.x); o[5] = f2bf(b.y); o[6] = f2bf(b.z); o[7] = f2bf(b.w);
    *(v8bf*)(dst + 8 * (size_t)i) = o;
}

// ---- float4 copy (init propagate buffer with the self term)
__global__ void gnn_copy4(const float4* __restrict__ src, float4* __restrict__ dst, int n4) {
    int i = blockIdx.x * blockDim.x + threadIdx.x;
    if (i < n4) dst[i] = src[i];
}

__global__ void gnn_zero4(float4* __restrict__ dst, int n4) {
    int i = blockIdx.x * blockDim.x + threadIdx.x;
    if (i < n4) dst[i] = make_float4(0.f, 0.f, 0.f, 0.f);
}

// ---- edge scatter-add: one wave32 per edge, 4 features per lane
__global__ void gnn_edge_scatter(const float* __restrict__ feat,
                                 const int* __restrict__ esrc,
                                 const int* __restrict__ edst,
                                 float* __restrict__ out, int n_edges) {
    int e    = blockIdx.x * 8 + (threadIdx.x >> 5);   // 8 waves per 256-thread block
    int lane = threadIdx.x & 31;
    if (e >= n_edges) return;
    int s = esrc[e], d = edst[e];
    float4 v = *(const float4*)(feat + (size_t)s * DFEAT + lane * 4);
    float* o = out + (size_t)d * DFEAT + lane * 4;
    atomicAdd(o + 0, v.x);
    atomicAdd(o + 1, v.y);
    atomicAdd(o + 2, v.z);
    atomicAdd(o + 3, v.w);
}

// ---- graph pooling: one wave32 per node, 4 features per lane
__global__ void gnn_pool(const float* __restrict__ h,
                         const int* __restrict__ idx,
                         float* __restrict__ pooled, int n_nodes) {
    int i    = blockIdx.x * 8 + (threadIdx.x >> 5);
    int lane = threadIdx.x & 31;
    if (i >= n_nodes) return;
    int g = idx[i];
    float4 v = *(const float4*)(h + (size_t)i * DFEAT + lane * 4);
    float* o = pooled + (size_t)g * DFEAT + lane * 4;
    atomicAdd(o + 0, v.x);
    atomicAdd(o + 1, v.y);
    atomicAdd(o + 2, v.z);
    atomicAdd(o + 3, v.w);
}

// ---- WMMA GEMM: C[rows x 128] = act(Abf[rows x 128](bf16) @ W(bf16, stored
//      TRANSPOSED as Wt[n][k]) + bias).  Layouts per CDNA5 ISA 7.12.2.
template <int RELU>
__global__ void gnn_gemm128(const __bf16* __restrict__ Abf,
                            const __bf16* __restrict__ Wt,
                            const float* __restrict__ bias,
                            float* __restrict__ C, int rows) {
    __shared__ __bf16 sW[DFEAT * LDS_STRIDE];      // ~34 KB of 320 KB/WGP

    // cooperative stage of Wt (32 KB) into LDS, 8 bf16 per thread per pass
    for (int t = threadIdx.x; t < DFEAT * (DFEAT / 8); t += 256) {
        int n = t >> 4;            // row 0..127
        int k = (t & 15) * 8;      // 16 chunks of 8 per row
        *(v8bf*)(&sW[n * LDS_STRIDE + k]) = *(const v8bf*)(Wt + n * DFEAT + k);
    }
    __syncthreads();               // all waves reach this before any strip-exit

    int tilesM = (rows + 15) >> 4;
    int w = blockIdx.x * 8 + (threadIdx.x >> 5);   // wave-uniform strip id
    if (w >= tilesM) return;                        // whole-wave exit: EXEC all-1s

    int lane = threadIdx.x & 31;
    int g    = lane >> 4;        // lane group 0/1
    int m16  = lane & 15;

    int arow = w * 16 + m16;
    if (arow > rows - 1) arow = rows - 1;           // clamp loads; stores guarded below
    const __bf16* Arow = Abf + (size_t)arow * DFEAT;

    v8f acc[8];
#pragma unroll
    for (int ct = 0; ct < 8; ++ct) acc[ct] = (v8f){};

#pragma unroll
    for (int kk = 0; kk < DFEAT; kk += 32) {
        // A fragment: e=0..7 -> K = kk+8g+e ; e=8..15 -> K = kk+16+8g+(e-8)
        v8bf a0 = *(const v8bf*)(Arow + kk + 8 * g);        // global_load_b128
        v8bf a1 = *(const v8bf*)(Arow + kk + 16 + 8 * g);   // global_load_b128
        v16bf af = __builtin_shufflevector(a0, a1,
            0, 1, 2, 3, 4, 5, 6, 7, 8, 9, 10, 11, 12, 13, 14, 15);

        // prefetch all 8 B fragments for this K-step from LDS (independent regs)
        v16bf bfr[8];
#pragma unroll
        for (int ct = 0; ct < 8; ++ct) {
            const __bf16* p = &sW[(ct * 16 + m16) * LDS_STRIDE + kk + 8 * g];
            v8bf b0 = *(const v8bf*)(p);         // ds_load_b128, 16B aligned
            v8bf b1 = *(const v8bf*)(p + 16);    // ds_load_b128, 16B aligned
            bfr[ct] = __builtin_shufflevector(b0, b1,
                0, 1, 2, 3, 4, 5, 6, 7, 8, 9, 10, 11, 12, 13, 14, 15);
        }

#pragma unroll
        for (int ct = 0; ct < 8; ++ct) {
            acc[ct] = __builtin_amdgcn_wmma_f32_16x16x32_bf16(
                /*neg_a=*/false, af, /*neg_b=*/false, bfr[ct],
                /*c_mod=*/(short)0, acc[ct], /*reuse_a=*/false, /*reuse_b=*/false);
        }
    }

    // C/D layout: VGPR r -> row M = r + 8g ; column N = m16 (per column tile)
#pragma unroll
    for (int ct = 0; ct < 8; ++ct) {
        int   col = ct * 16 + m16;
        float bv  = bias[col];
#pragma unroll
        for (int r = 0; r < 8; ++r) {
            int outRow = w * 16 + r + 8 * g;
            if (outRow < rows) {
                float v = acc[ct][r] + bv;
                if (RELU) v = v > 0.f ? v : 0.f;
                C[(size_t)outRow * DFEAT + col] = v;
            }
        }
    }
}

// ---- head: out = log_softmax(zg @ W4 + b4); one thread per graph (500x16)
__global__ void gnn_head(const float* __restrict__ zg,
                         const float* __restrict__ W4,
                         const float* __restrict__ b4,
                         float* __restrict__ out, int n_graphs) {
    int gidx = blockIdx.x * blockDim.x + threadIdx.x;
    if (gidx >= n_graphs) return;
    const float* zr = zg + (size_t)gidx * DFEAT;
    float logits[N_CLASS];
#pragma unroll
    for (int c = 0; c < N_CLASS; ++c) logits[c] = b4[c];
    for (int k = 0; k < DFEAT; ++k) {
        float zv = zr[k];
        const float* wrow = W4 + (size_t)k * N_CLASS;
#pragma unroll
        for (int c = 0; c < N_CLASS; ++c) logits[c] = fmaf(zv, wrow[c], logits[c]);
    }
    float mx = logits[0];
#pragma unroll
    for (int c = 1; c < N_CLASS; ++c) mx = fmaxf(mx, logits[c]);
    float se = 0.f;
#pragma unroll
    for (int c = 0; c < N_CLASS; ++c) se += __expf(logits[c] - mx);
    float lse = mx + __logf(se);
#pragma unroll
    for (int c = 0; c < N_CLASS; ++c) out[(size_t)gidx * N_CLASS + c] = logits[c] - lse;
}

extern "C" void kernel_launch(void* const* d_in, const int* in_sizes, int n_in,
                              void* d_out, int out_size, void* d_ws, size_t ws_size,
                              hipStream_t stream) {
    const float* x    = (const float*)d_in[0];
    const int*   esrc = (const int*)d_in[1];
    const int*   edst = (const int*)d_in[2];
    const int*   idx  = (const int*)d_in[3];
    const float* W1 = (const float*)d_in[4];
    const float* b1 = (const float*)d_in[5];
    const float* W2 = (const float*)d_in[6];
    const float* b2 = (const float*)d_in[7];
    const float* W3 = (const float*)d_in[8];
    const float* b3 = (const float*)d_in[9];
    const float* W4 = (const float*)d_in[10];
    const float* b4 = (const float*)d_in[11];
    float* out = (float*)d_out;

    // workspace layout (all blocks 256B-aligned by construction)
    char* ws = (char*)d_ws;
    size_t featBytes   = (size_t)N_NODES * DFEAT * sizeof(float);   // 25.6 MB
    size_t featBfBytes = (size_t)N_NODES * DFEAT * sizeof(__bf16);  // 12.8 MB
    float*  bufA     = (float*)ws;        ws += featBytes;           // p1 / p2
    float*  bufB     = (float*)ws;        ws += featBytes;           // z1 / h
    __bf16* actBf    = (__bf16*)ws;       ws += featBfBytes;         // bf16 activations
    float*  pooled   = (float*)ws;        ws += (size_t)N_GRAPHS * DFEAT * sizeof(float);
    float*  zg       = (float*)ws;        ws += (size_t)N_GRAPHS * DFEAT * sizeof(float);
    __bf16* pooledBf = (__bf16*)ws;       ws += (size_t)N_GRAPHS * DFEAT * sizeof(__bf16);
    __bf16* Wt1      = (__bf16*)ws;       ws += (size_t)DFEAT * DFEAT * sizeof(__bf16);
    __bf16* Wt2      = (__bf16*)ws;       ws += (size_t)DFEAT * DFEAT * sizeof(__bf16);
    __bf16* Wt3     = (__bf16*)ws;

    int n4feat   = N_NODES * DFEAT / 4;      // 1.6M float4
    int n8feat   = N_NODES * DFEAT / 8;      // 800K cvt threads
    int n4pool   = N_GRAPHS * DFEAT / 4;
    int n8pool   = N_GRAPHS * DFEAT / 8;
    int cpBlocks = (n4feat + 255) / 256;
    int cvBlocks = (n8feat + 255) / 256;
    int edgeBlk  = (N_EDGES + 7) / 8;        // 1 wave per edge, 8 waves/block
    int poolBlk  = (N_NODES + 7) / 8;
    int stripsBig = (N_NODES + 15) / 16;     // 3125 strips (waves)
    int stripsSm  = (N_GRAPHS + 15) / 16;    // 32 strips
    int gemmBlkBig = (stripsBig + 7) / 8;    // 8 waves per block
    int gemmBlkSm  = (stripsSm + 7) / 8;

    // weight prep (bf16, transposed)
    gnn_prep_w<<<64, 256, 0, stream>>>(W1, Wt1);
    gnn_prep_w<<<64, 256, 0, stream>>>(W2, Wt2);
    gnn_prep_w<<<64, 256, 0, stream>>>(W3, Wt3);

    // p1 = x + A x
    gnn_copy4<<<cpBlocks, 256, 0, stream>>>((const float4*)x, (float4*)bufA, n4feat);
    gnn_edge_scatter<<<edgeBlk, 256, 0, stream>>>(x, esrc, edst, bufA, N_EDGES);
    // z1 = relu(p1 @ W1 + b1)
    gnn_cvt_bf16<<<cvBlocks, 256, 0, stream>>>(bufA, actBf, n8feat);
    gnn_gemm128<1><<<gemmBlkBig, 256, 0, stream>>>(actBf, Wt1, b1, bufB, N_NODES);
    // p2 = z1 + A z1   (reuse bufA)
    gnn_copy4<<<cpBlocks, 256, 0, stream>>>((const float4*)bufB, (float4*)bufA, n4feat);
    gnn_edge_scatter<<<edgeBlk, 256, 0, stream>>>(bufB, esrc, edst, bufA, N_EDGES);
    // h = p2 @ W2 + b2  (reuse bufB)
    gnn_cvt_bf16<<<cvBlocks, 256, 0, stream>>>(bufA, actBf, n8feat);
    gnn_gemm128<0><<<gemmBlkBig, 256, 0, stream>>>(actBf, Wt2, b2, bufB, N_NODES);
    // pooled = segment_sum(h, idx)
    gnn_zero4<<<(n4pool + 255) / 256, 256, 0, stream>>>((float4*)pooled, n4pool);
    gnn_pool<<<poolBlk, 256, 0, stream>>>(bufB, idx, pooled, N_NODES);
    // zg = relu(pooled @ W3 + b3)
    gnn_cvt_bf16<<<(n8pool + 255) / 256, 256, 0, stream>>>(pooled, pooledBf, n8pool);
    gnn_gemm128<1><<<gemmBlkSm, 256, 0, stream>>>(pooledBf, Wt3, b3, zg, N_GRAPHS);
    // out = log_softmax(zg @ W4 + b4)
    gnn_head<<<(N_GRAPHS + 255) / 256, 256, 0, stream>>>(zg, W4, b4, out, N_GRAPHS);
}
